// WindowAttention_2508260901408
// MI455X (gfx1250) — matrix operand order repaired and checked by reading, not compiled, not run on hardware
//
#include <hip/hip_runtime.h>
#include <cstdint>
#include <cstddef>

// ---------------------------------------------------------------------------
// Types for CDNA5 WMMA / TDM
// ---------------------------------------------------------------------------
typedef __attribute__((ext_vector_type(16))) __bf16 v16bf;
typedef __attribute__((ext_vector_type(8)))  float  v8f;
typedef __attribute__((ext_vector_type(4))) unsigned int tdm_g0_t;
typedef __attribute__((ext_vector_type(8))) int          tdm_g1_t;
typedef __attribute__((ext_vector_type(4))) int          tdm_g2_t;

#if defined(__has_builtin)
#if __has_builtin(__builtin_amdgcn_tensor_load_to_lds) && \
    __has_builtin(__builtin_amdgcn_s_wait_tensorcnt)
#define USE_TDM 1
#endif
#endif
#ifndef USE_TDM
#define USE_TDM 0
#endif

// ---------------------------------------------------------------------------
// Problem constants (match reference)
// ---------------------------------------------------------------------------
#define BATCH   16
#define PW      64
#define DIM     512
#define HEADS   8
#define DH      64
#define WS      8
#define WS2     64
#define INNER   512            // HEADS*DH
#define NQKV    1536           // 3*INNER
#define NTOK    (BATCH * PW * PW)   // 65536
#define ATTN_SCALE 0.125f      // DH^-0.5

// ---------------------------------------------------------------------------
// Workspace layout (bytes, all 256-aligned)
// ---------------------------------------------------------------------------
static constexpr size_t OFF_XB   = 0;                          // x bf16        64 MB
static constexpr size_t OFF_WQKV = 67108864;                   // w_qkv bf16   1.5 MB
static constexpr size_t OFF_WPRO = OFF_WQKV + 1572864;         // w_pro bf16   4.5 MB
static constexpr size_t OFF_WOUT = OFF_WPRO + 4718592;         // w_out bf16   0.5 MB
static constexpr size_t OFF_H1   = OFF_WOUT + 524288;          // h1 bf16      192 MB (reused: attn out 2x64MB)
static constexpr size_t OFF_QKV  = OFF_H1 + 201326592;         // qkv bf16     192 MB
// attn shift outputs alias OFF_H1 / OFF_H1+64MB; avg aliases OFF_XB.

// ---------------------------------------------------------------------------
// LDS fragment loaders per CDNA5 WMMA VGPR layouts (wave32, 16-bit data).
// A 16x32 (MxK): lane 0-15 -> K group base 0, lane 16-31 -> base 8;
//   vgpr j<4: K = kb+2j,kb+2j+1 ; j>=4: K = 16+kb+2(j-4),+1
// B 32x16 (KxN): same K order as A (lane 0-15 -> base 0, lane 16-31 -> base 8)
// Stored so that K is the contiguous LDS dimension -> ds_load_b32 pairs.
// ---------------------------------------------------------------------------
__device__ __forceinline__ v16bf load_a_frag(const __bf16* p, int stride, int row0, int k0) {
    const int lane = threadIdx.x & 31;
    const int m  = row0 + (lane & 15);
    const int kb = (lane & 16) ? 8 : 0;
    const __bf16* rp = p + m * stride + k0;
    v16bf f;
    uint32_t* u = reinterpret_cast<uint32_t*>(&f);
#pragma unroll
    for (int j = 0; j < 8; ++j) {
        const int k = (j < 4) ? (kb + 2 * j) : (16 + kb + 2 * (j - 4));
        u[j] = *reinterpret_cast<const uint32_t*>(rp + k);
    }
    return f;
}

__device__ __forceinline__ v16bf load_b_frag(const __bf16* p, int stride, int col0, int k0) {
    const int lane = threadIdx.x & 31;
    const int n  = col0 + (lane & 15);
    const int kb = (lane & 16) ? 8 : 0;
    const __bf16* rp = p + n * stride + k0;
    v16bf f;
    uint32_t* u = reinterpret_cast<uint32_t*>(&f);
#pragma unroll
    for (int j = 0; j < 8; ++j) {
        const int k = (j < 4) ? (kb + 2 * j) : (16 + kb + 2 * (j - 4));
        u[j] = *reinterpret_cast<const uint32_t*>(rp + k);
    }
    return f;
}

__device__ __forceinline__ float gelu_exact(float v) {
    return 0.5f * v * (1.0f + erff(v * 0.70710678118654752f));
}

// ---------------------------------------------------------------------------
// f32 -> bf16 conversion (4 elems / thread)
// ---------------------------------------------------------------------------
__global__ void cvt_f32_bf16(const float* __restrict__ in, __bf16* __restrict__ out, int n) {
    int i = (blockIdx.x * blockDim.x + threadIdx.x) * 4;
#pragma unroll
    for (int j = 0; j < 4; ++j) {
        if (i + j < n) out[i + j] = (__bf16)in[i + j];
    }
}

// ---------------------------------------------------------------------------
// Average the two shift outputs: avg = 0.5*(a+b), bf16
// ---------------------------------------------------------------------------
__global__ void combine_avg(const __bf16* __restrict__ a, const __bf16* __restrict__ b,
                            __bf16* __restrict__ out, int n) {
    int i = (blockIdx.x * blockDim.x + threadIdx.x) * 4;
#pragma unroll
    for (int j = 0; j < 4; ++j) {
        if (i + j < n) out[i + j] = (__bf16)(0.5f * ((float)a[i + j] + (float)b[i + j]));
    }
}

// ---------------------------------------------------------------------------
// Tiled bf16 WMMA GEMM:  C(MxN) = act(A(MxK) * B(KxN) + bias)
// 256 threads = 8 waves; tile BM=128 BN=128 BK=32; wave -> 32x64 sub-tile.
// A tile staged by the Tensor Data Mover (wave 0 issues a 2-D descriptor,
// TDM padding reproduces the 40-half LDS row stride); B tile staged by all
// threads with an in-LDS transpose so K stays contiguous for B fragments.
// ---------------------------------------------------------------------------
template <bool GELU_ACT, bool OUT_F32>
__global__ __launch_bounds__(256)
void gemm_bf16_wmma(const __bf16* __restrict__ A, const __bf16* __restrict__ Bm,
                    const float* __restrict__ bias, void* __restrict__ Cout,
                    int M, int N, int K) {
    constexpr int BM = 128, BN = 128, BK = 32, LDT = BK + 8;
    __shared__ __bf16 As[BM][LDT];
    __shared__ __bf16 Bt[BN][LDT];   // stored transposed: [n][k]

    const int tid  = threadIdx.x;
    const int lane = tid & 31;
    const int wave = tid >> 5;
    const int m0 = blockIdx.y * BM;
    const int n0 = blockIdx.x * BN;
    const int wm = (wave >> 1) * 32;   // wave row offset in tile
    const int wn = (wave & 1) * 64;    // wave col offset in tile

#if USE_TDM
    const uint64_t Abase  = (uint64_t)(uintptr_t)A;
    const uint32_t ldsA   = (uint32_t)(uintptr_t)(&As[0][0]);  // LDS aperture: offset in [31:0]
#endif

    v8f acc[2][4];
#pragma unroll
    for (int mi = 0; mi < 2; ++mi)
#pragma unroll
        for (int ni = 0; ni < 4; ++ni) acc[mi][ni] = (v8f)0.0f;

    for (int k0 = 0; k0 < K; k0 += BK) {
        __syncthreads();
#if USE_TDM
        // --- stage A tile via TDM: 128x32 bf16 tile of the MxK tensor -------
        if (wave == 0) {
            const uint64_t ga = Abase + (((uint64_t)m0 * (uint64_t)K + (uint64_t)k0) << 1);
            tdm_g0_t g0;
            g0.x = 1u;                                   // count=1, no gather
            g0.y = ldsA;                                 // lds_addr
            g0.z = (uint32_t)ga;                         // global_addr[31:0]
            g0.w = (uint32_t)((ga >> 32) & 0x01FFFFFFu) | (2u << 30);  // addr[56:32] | type=2
            tdm_g1_t g1;
            // data_size=1 (2B) | pad_enable | pad_interval=3 (16 dw) | pad_amount=3 (4 dw)
            g1[0] = (int)((1u << 16) | (1u << 20) | (3u << 22) | (3u << 25));
            g1[1] = (int)(((uint32_t)K & 0xFFFFu) << 16);                       // tensor_dim0 lo
            g1[2] = (int)((((uint32_t)K >> 16) & 0xFFFFu) |
                          (((uint32_t)M & 0xFFFFu) << 16));                     // dim0 hi | dim1 lo
            g1[3] = (int)((((uint32_t)M >> 16) & 0xFFFFu) | (32u << 16));       // dim1 hi | tile_dim0
            g1[4] = 128;                                                        // tile_dim1
            g1[5] = K;                                                          // dim0_stride lo
            g1[6] = 0;
            g1[7] = 0;
            tdm_g2_t gz  = {0, 0, 0, 0};
            tdm_g1_t gz8 = {0, 0, 0, 0, 0, 0, 0, 0};
            __builtin_amdgcn_tensor_load_to_lds(g0, g1, gz, gz, gz8, 0);
        }
#else
        // --- fallback: stage A tile with vector copies ----------------------
        {
            const int row = tid >> 1;
            const int cb  = (tid & 1) * 16;
            const __bf16* g = A + (size_t)(m0 + row) * K + k0 + cb;
            uint4 r0 = *reinterpret_cast<const uint4*>(g);
            uint4 r1 = *reinterpret_cast<const uint4*>(g + 8);
            *reinterpret_cast<uint4*>(&As[row][cb])     = r0;
            *reinterpret_cast<uint4*>(&As[row][cb + 8]) = r1;
        }
#endif
        // --- stage B tile transposed: read [k][n] contiguous, scatter Bt[n][k]
        {
            const int kr = tid >> 3;
            const int nb = (tid & 7) * 16;
            const __bf16* g = Bm + (size_t)(k0 + kr) * N + n0 + nb;
            __bf16 tmp[16];
            *reinterpret_cast<uint4*>(tmp)     = *reinterpret_cast<const uint4*>(g);
            *reinterpret_cast<uint4*>(tmp + 8) = *reinterpret_cast<const uint4*>(g + 8);
#pragma unroll
            for (int c = 0; c < 16; ++c) Bt[nb + c][kr] = tmp[c];
        }
#if USE_TDM
        if (wave == 0) __builtin_amdgcn_s_wait_tensorcnt(0);
#endif
        __syncthreads();

        v16bf af[2];
#pragma unroll
        for (int mi = 0; mi < 2; ++mi)
            af[mi] = load_a_frag(&As[0][0], LDT, wm + mi * 16, 0);
#pragma unroll
        for (int ni = 0; ni < 4; ++ni) {
            v16bf bf = load_b_frag(&Bt[0][0], LDT, wn + ni * 16, 0);
#pragma unroll
            for (int mi = 0; mi < 2; ++mi) {
                acc[mi][ni] = __builtin_amdgcn_wmma_f32_16x16x32_bf16(
                    false, af[mi], false, bf, (short)0, acc[mi][ni], false, false);
            }
        }
    }

    // --- epilogue: C/D layout -> vgpr r: M = r (lanes 0-15) or r+8 (lanes 16-31)
#pragma unroll
    for (int mi = 0; mi < 2; ++mi) {
#pragma unroll
        for (int ni = 0; ni < 4; ++ni) {
#pragma unroll
            for (int r = 0; r < 8; ++r) {
                const int row = m0 + wm + mi * 16 + ((lane & 16) ? (r + 8) : r);
                const int col = n0 + wn + ni * 16 + (lane & 15);
                float v = acc[mi][ni][r];
                if (bias) v += bias[col];
                if (GELU_ACT) v = gelu_exact(v);
                if (OUT_F32)
                    reinterpret_cast<float*>(Cout)[(size_t)row * N + col] = v;
                else
                    reinterpret_cast<__bf16*>(Cout)[(size_t)row * N + col] = (__bf16)v;
            }
        }
    }
}

// ---------------------------------------------------------------------------
// Window attention: one wave32 per (window, head); blockIdx.y = shift index.
// Cyclic roll folded into gather/scatter indices. All matmuls via WMMA bf16.
// ---------------------------------------------------------------------------
__global__ __launch_bounds__(32)
void window_attn_wmma(const __bf16* __restrict__ qkv,      // [NTOK][1536] bf16
                      const float* __restrict__ rpb_table, // [225][8]
                      const int*   __restrict__ rel_index, // [64][64]
                      const float* __restrict__ attn_masks,// [2][64][64][64]
                      __bf16* __restrict__ out0,           // [NTOK][512]
                      __bf16* __restrict__ out1) {
    constexpr int LDH = DH + 4;    // 68: bf16 tile stride
    constexpr int LDS_S = WS2 + 2; // 66: f32 score stride
    __shared__ __bf16 Qs[WS2][LDH];   // Q, later reused as P (bf16)
    __shared__ __bf16 Ks[WS2][LDH];   // K: [token][channel]
    __shared__ __bf16 Vt[DH][LDH];    // V transposed: [channel][token]
    __shared__ float  biasS[WS2][WS2];
    __shared__ float  Ssc[WS2][LDS_S];

    const int tid  = threadIdx.x;
    const int lane = tid & 31;
    const int h    = blockIdx.x & 7;
    const int win  = blockIdx.x >> 3;
    const int b    = win >> 6;
    const int wi   = win & 63;
    const int wy   = wi >> 3;
    const int wx   = wi & 7;
    const int s    = blockIdx.y;       // 0 or 1
    const int shift = s * 4;

    // ---- stage Q,K,V head slices (with shift roll) + bias tile -------------
    for (int it = 0; it < 128; ++it) {
        const int e   = it * 32 + tid;
        const int tok = e >> 6;
        const int c   = e & 63;
        const int ty = tok >> 3, tx = tok & 7;
        const int y = (wy * WS + ty + shift) & (PW - 1);
        const int x = (wx * WS + tx + shift) & (PW - 1);
        const size_t base = ((size_t)((b * PW + y) * PW + x)) * NQKV + h * DH + c;
        Qs[tok][c] = qkv[base];
        Ks[tok][c] = qkv[base + INNER];
        Vt[c][tok] = qkv[base + 2 * INNER];
        // bias = rpb[rel_index[i,j], h] + mask[s, wi, i, j]
        const int i = tok, j = c;
        biasS[i][j] = rpb_table[rel_index[i * WS2 + j] * HEADS + h] +
                      attn_masks[(((size_t)s * 64 + wi) * WS2 + i) * WS2 + j];
    }
    __syncthreads();

    // ---- S = Q K^T * scale + bias  (64x64, K-dim 64 = 2 wmma steps) --------
#pragma unroll
    for (int mi = 0; mi < 4; ++mi) {
        v8f acc[4];
#pragma unroll
        for (int ni = 0; ni < 4; ++ni) acc[ni] = (v8f)0.0f;
#pragma unroll
        for (int kk = 0; kk < 2; ++kk) {
            v16bf a = load_a_frag(&Qs[0][0], LDH, mi * 16, kk * 32);
#pragma unroll
            for (int ni = 0; ni < 4; ++ni) {
                v16bf bfr = load_b_frag(&Ks[0][0], LDH, ni * 16, kk * 32);
                acc[ni] = __builtin_amdgcn_wmma_f32_16x16x32_bf16(
                    false, a, false, bfr, (short)0, acc[ni], false, false);
            }
        }
#pragma unroll
        for (int ni = 0; ni < 4; ++ni) {
#pragma unroll
            for (int r = 0; r < 8; ++r) {
                const int i = mi * 16 + ((lane & 16) ? (r + 8) : r);
                const int j = ni * 16 + (lane & 15);
                Ssc[i][j] = acc[ni][r] * ATTN_SCALE + biasS[i][j];
            }
        }
    }
    __syncthreads();

    // ---- row softmax (2 rows per lane), write P (bf16) over Q buffer -------
#pragma unroll
    for (int rr = 0; rr < 2; ++rr) {
        const int i = tid * 2 + rr;
        float mx = -3.0e38f;
#pragma unroll
        for (int j = 0; j < WS2; ++j) mx = fmaxf(mx, Ssc[i][j]);
        float sum = 0.0f;
        float e[WS2];
#pragma unroll
        for (int j = 0; j < WS2; ++j) { e[j] = expf(Ssc[i][j] - mx); sum += e[j]; }
        const float inv = 1.0f / sum;
#pragma unroll
        for (int j = 0; j < WS2; ++j) Qs[i][j] = (__bf16)(e[j] * inv);
    }
    __syncthreads();

    // ---- O = P V  (64 tokens x 64 channels), scatter with inverse roll -----
    __bf16* outp = s ? out1 : out0;
#pragma unroll
    for (int mi = 0; mi < 4; ++mi) {
        v8f acc[4];
#pragma unroll
        for (int ni = 0; ni < 4; ++ni) acc[ni] = (v8f)0.0f;
#pragma unroll
        for (int kk = 0; kk < 2; ++kk) {
            v16bf a = load_a_frag(&Qs[0][0], LDH, mi * 16, kk * 32);  // P
#pragma unroll
            for (int ni = 0; ni < 4; ++ni) {
                v16bf bfr = load_b_frag(&Vt[0][0], LDH, ni * 16, kk * 32);
                acc[ni] = __builtin_amdgcn_wmma_f32_16x16x32_bf16(
                    false, a, false, bfr, (short)0, acc[ni], false, false);
            }
        }
#pragma unroll
        for (int ni = 0; ni < 4; ++ni) {
#pragma unroll
            for (int r = 0; r < 8; ++r) {
                const int i = mi * 16 + ((lane & 16) ? (r + 8) : r);   // token
                const int n = ni * 16 + (lane & 15);                   // channel
                const int ty = i >> 3, tx = i & 7;
                const int y = (wy * WS + ty + shift) & (PW - 1);
                const int x = (wx * WS + tx + shift) & (PW - 1);
                const size_t pos = (size_t)((b * PW + y) * PW + x);
                outp[pos * INNER + h * DH + n] = (__bf16)acc[ni][r];
            }
        }
    }
}

// ---------------------------------------------------------------------------
// Host launch
// ---------------------------------------------------------------------------
extern "C" void kernel_launch(void* const* d_in, const int* in_sizes, int n_in,
                              void* d_out, int out_size, void* d_ws, size_t ws_size,
                              hipStream_t stream) {
    (void)in_sizes; (void)n_in; (void)out_size; (void)ws_size;
    const float* x      = (const float*)d_in[0];
    const float* w_qkv  = (const float*)d_in[1];
    const float* w_pro  = (const float*)d_in[2];
    const float* b_pro  = (const float*)d_in[3];
    const float* w_out  = (const float*)d_in[4];
    const float* b_out  = (const float*)d_in[5];
    const float* rpb    = (const float*)d_in[6];
    const float* masks  = (const float*)d_in[7];
    const int*   relidx = (const int*)d_in[8];
    float* out = (float*)d_out;

    char* ws = (char*)d_ws;
    __bf16* xb    = (__bf16*)(ws + OFF_XB);
    __bf16* wqkvb = (__bf16*)(ws + OFF_WQKV);
    __bf16* wprob = (__bf16*)(ws + OFF_WPRO);
    __bf16* woutb = (__bf16*)(ws + OFF_WOUT);
    __bf16* h1b   = (__bf16*)(ws + OFF_H1);
    __bf16* qkvb  = (__bf16*)(ws + OFF_QKV);
    __bf16* attn0 = (__bf16*)(ws + OFF_H1);                 // alias (h1 dead after GEMM2)
    __bf16* attn1 = (__bf16*)(ws + OFF_H1 + 67108864);
    __bf16* avgb  = (__bf16*)(ws + OFF_XB);                 // alias (x bf16 dead after GEMM1)

    // 1) convert inputs to bf16
    const int nx = NTOK * DIM;                 // 33.5M
    cvt_f32_bf16<<<nx / 1024, 256, 0, stream>>>(x, xb, nx);
    cvt_f32_bf16<<<(DIM * NQKV + 1023) / 1024, 256, 0, stream>>>(w_qkv, wqkvb, DIM * NQKV);
    cvt_f32_bf16<<<(NQKV * NQKV + 1023) / 1024, 256, 0, stream>>>(w_pro, wprob, NQKV * NQKV);
    cvt_f32_bf16<<<(INNER * DIM + 1023) / 1024, 256, 0, stream>>>(w_out, woutb, INNER * DIM);

    // 2) h1 = gelu(x @ w_qkv)           [65536 x 1536], bf16 out
    gemm_bf16_wmma<true, false><<<dim3(NQKV / 128, NTOK / 128), 256, 0, stream>>>(
        xb, wqkvb, nullptr, h1b, NTOK, NQKV, DIM);

    // 3) qkv = h1 @ w_pro + b_pro       [65536 x 1536], bf16 out
    gemm_bf16_wmma<false, false><<<dim3(NQKV / 128, NTOK / 128), 256, 0, stream>>>(
        h1b, wprob, b_pro, qkvb, NTOK, NQKV, NQKV);

    // 4) window attention for both shifts (one wave per window*head)
    window_attn_wmma<<<dim3(BATCH * 64 * HEADS, 2), 32, 0, stream>>>(
        qkvb, rpb, relidx, masks, attn0, attn1);

    // 5) avg of the two shifts
    combine_avg<<<nx / 1024, 256, 0, stream>>>(attn0, attn1, avgb, nx);

    // 6) out = avg @ w_out + b_out      [65536 x 512], f32 out
    gemm_bf16_wmma<false, true><<<dim3(DIM / 128, NTOK / 128), 256, 0, stream>>>(
        avgb, woutb, b_out, out, NTOK, DIM, DIM);
}
